// CLFMv2_NoSpatialCoords_39436389712133
// MI455X (gfx1250) — matrix operand order repaired and checked by reading, not compiled, not run on hardware
//
#include <hip/hip_runtime.h>
#include <hip/hip_bf16.h>

// ---------------------------------------------------------------------------
// CLFMv2 forward on MI455X (gfx1250, wave32).
//
// Pipeline (all on `stream`, graph-capture safe):
//   1. adj_kernel     : adj = softmax(relu(E1 @ E2^T)) -> bf16 (32 MB, L2-resident)
//   2. encoder_kernel : per-(b,n) MLP encoder + temporal embeddings
//                       -> field f32, state=0, Ft bf16 (transposed field)
//   3. 4x { gemm_adj  : Yt = adj @ Ft^T via v_wmma_f32_16x16x32_bf16
//           pde_step  : pointwise PDE + GRU update, rewrites field/state/Ft }
//   4. decoder_kernel : per-(b,n) MLP decoder -> d_out (B, OUT, N, 1) f32
// ---------------------------------------------------------------------------

#define B_   32
#define L_   12
#define N_   4096
#define FD_  32
#define HD_  64
#define OUT_ 12
#define STEPS_ 4
#define BFD_ (B_ * FD_)   // 1024 columns of the folded GEMM

typedef __attribute__((ext_vector_type(16))) __bf16 v16bf;
typedef __attribute__((ext_vector_type(8)))  float  v8f;

union PackA { uint4 q[2]; v16bf v; };

__device__ __forceinline__ unsigned short f2bf(float x) {
    unsigned int u = __float_as_uint(x);
    unsigned int r = (u + 0x7FFFu + ((u >> 16) & 1u)) >> 16;
    return (unsigned short)r;
}

__device__ __forceinline__ float sigmf(float x) { return 1.0f / (1.0f + __expf(-x)); }

// ---------------------------------------------------------------------------
// adj = softmax(relu(E1 @ E2^T), axis=-1) stored as bf16 row-major N x N.
// One block per row; 256 threads x 16 columns each.
// ---------------------------------------------------------------------------
__global__ __launch_bounds__(256) void adj_kernel(const float* __restrict__ E1,
                                                  const float* __restrict__ E2,
                                                  unsigned short* __restrict__ adjbf) {
    __shared__ float red[256];
    const int n = blockIdx.x;
    const int t = threadIdx.x;

    float e1[16];
#pragma unroll
    for (int k = 0; k < 16; ++k) e1[k] = E1[n * 16 + k];

    float d[16];
    float lmax = -1e30f;
#pragma unroll
    for (int i = 0; i < 16; ++i) {
        const int m = t + (i << 8);
        float a = 0.0f;
#pragma unroll
        for (int k = 0; k < 16; ++k) a = fmaf(e1[k], E2[m * 16 + k], a);
        a = fmaxf(a, 0.0f);
        d[i] = a;
        lmax = fmaxf(lmax, a);
    }
    red[t] = lmax;
    __syncthreads();
    for (int off = 128; off > 0; off >>= 1) {
        if (t < off) red[t] = fmaxf(red[t], red[t + off]);
        __syncthreads();
    }
    const float rmax = red[0];
    __syncthreads();

    float lsum = 0.0f;
#pragma unroll
    for (int i = 0; i < 16; ++i) {
        d[i] = __expf(d[i] - rmax);
        lsum += d[i];
    }
    red[t] = lsum;
    __syncthreads();
    for (int off = 128; off > 0; off >>= 1) {
        if (t < off) red[t] += red[t + off];
        __syncthreads();
    }
    const float inv = 1.0f / red[0];
#pragma unroll
    for (int i = 0; i < 16; ++i) {
        const int m = t + (i << 8);
        adjbf[(size_t)n * N_ + m] = f2bf(d[i] * inv);
    }
}

// ---------------------------------------------------------------------------
// Residual MLP block: h = h + relu(h@W1 + b1) @ W2 + b2   (64-wide)
// Streaming over hidden unit j keeps h[] / acc[] fully register-resident.
// ---------------------------------------------------------------------------
__device__ __forceinline__ void mlp64(float* h,
                                      const float* __restrict__ f1W, const float* __restrict__ f1B,
                                      const float* __restrict__ f2W, const float* __restrict__ f2B) {
    float acc[HD_];
#pragma unroll
    for (int k = 0; k < HD_; ++k) acc[k] = h[k] + f2B[k];
    for (int j = 0; j < HD_; ++j) {
        float a = f1B[j];
#pragma unroll
        for (int i = 0; i < HD_; ++i) a = fmaf(h[i], f1W[i * HD_ + j], a);
        a = fmaxf(a, 0.0f);
#pragma unroll
        for (int k = 0; k < HD_; ++k) acc[k] = fmaf(a, f2W[j * HD_ + k], acc[k]);
    }
#pragma unroll
    for (int k = 0; k < HD_; ++k) h[k] = acc[k];
}

// ---------------------------------------------------------------------------
// Encoder: one thread per (b, n).
// ---------------------------------------------------------------------------
__global__ __launch_bounds__(256) void encoder_kernel(
    const float* __restrict__ hist,
    const float* __restrict__ eiW, const float* __restrict__ eiB,
    const float* __restrict__ m1aW, const float* __restrict__ m1aB,
    const float* __restrict__ m1bW, const float* __restrict__ m1bB,
    const float* __restrict__ m2aW, const float* __restrict__ m2aB,
    const float* __restrict__ m2bW, const float* __restrict__ m2bB,
    const float* __restrict__ eoW, const float* __restrict__ eoB,
    const float* __restrict__ tidE, const float* __restrict__ diwE,
    const float* __restrict__ tfW, const float* __restrict__ tfB,
    float* __restrict__ field, float* __restrict__ state,
    unsigned short* __restrict__ Ft) {
    const int idx = blockIdx.x * blockDim.x + threadIdx.x;  // b*N + n
    const int b = idx >> 12;
    const int n = idx & (N_ - 1);

    float xf[L_];
#pragma unroll
    for (int l = 0; l < L_; ++l)
        xf[l] = hist[((size_t)(b * L_ + l) * N_ + n) * 3 + 0];

    float h[HD_];
#pragma unroll
    for (int j = 0; j < HD_; ++j) {
        float a = eiB[j];
#pragma unroll
        for (int l = 0; l < L_; ++l) a = fmaf(xf[l], eiW[l * HD_ + j], a);
        h[j] = a;
    }

    mlp64(h, m1aW, m1aB, m1bW, m1bB);
    mlp64(h, m2aW, m2aB, m2bW, m2bB);

    float fld[FD_];
#pragma unroll
    for (int k = 0; k < FD_; ++k) fld[k] = eoB[k];
#pragma unroll
    for (int i = 0; i < HD_; ++i) {
        const float hi = h[i];
#pragma unroll
        for (int k = 0; k < FD_; ++k) fld[k] = fmaf(hi, eoW[i * FD_ + k], fld[k]);
    }

    // temporal embedding at last timestep
    const float tv = hist[((size_t)(b * L_ + (L_ - 1)) * N_ + n) * 3 + 1];
    const float dv = hist[((size_t)(b * L_ + (L_ - 1)) * N_ + n) * 3 + 2];
    int tid = (int)(tv * 288.0f); tid = tid < 0 ? 0 : (tid > 287 ? 287 : tid);
    int diw = (int)(dv * 7.0f);   diw = diw < 0 ? 0 : (diw > 6 ? 6 : diw);

    float tmp[32];
#pragma unroll
    for (int k = 0; k < 16; ++k) {
        tmp[k]      = tidE[tid * 16 + k];
        tmp[16 + k] = diwE[diw * 16 + k];
    }
#pragma unroll
    for (int i = 0; i < 32; ++i) {
        const float ti = tmp[i];
#pragma unroll
        for (int k = 0; k < FD_; ++k) fld[k] = fmaf(ti, tfW[i * FD_ + k], fld[k]);
    }
#pragma unroll
    for (int k = 0; k < FD_; ++k) fld[k] += tfB[k];

    const size_t base = (size_t)idx * FD_;
#pragma unroll
    for (int k = 0; k < FD_; ++k) {
        field[base + k] = fld[k];
        state[base + k] = 0.0f;
        Ft[(size_t)(b * FD_ + k) * N_ + n] = f2bf(fld[k]);
    }
}

// ---------------------------------------------------------------------------
// GEMM: Yt[col][n] = sum_m adj[n][m] * Ft[col][m]
//   M = 4096 (n), N = 1024 (col = b*32+f), K = 4096 (m), bf16 -> f32 WMMA.
// One wave computes a 32(M) x 64(N) strip:
//   2 A fragments (rows nBase..+15, nBase+16..+31) reused across 4 col tiles,
//   4 B fragments reused across both row tiles -> 12 loads per 8 WMMAs.
// Fragment loads use one contiguous k order for A and B (not the ISA table order):
//   A 16x32 bf16: lane L -> row L%16, K chunk 16*(L/16)..+15 (contiguous)
//   B 32x16 bf16: lane L -> col L%16, K chunk 16*(L/16)..+15 (contiguous in Ft)
//   C 16x16 f32 : lane L, vgpr r -> M = r + 8*(L/16), N = L%16
// ---------------------------------------------------------------------------
__global__ __launch_bounds__(128) void gemm_adj(const unsigned short* __restrict__ adjbf,
                                                const unsigned short* __restrict__ Ft,
                                                float* __restrict__ Yt) {
    const int wid  = threadIdx.x >> 5;
    const int lane = threadIdx.x & 31;
    const int gw   = blockIdx.x * 4 + wid;     // 2048 waves total
    const int mTile  = gw & 127;               // 128 M-tiles of 32 rows
    const int cGroup = gw >> 7;                // 16 groups of 64 columns
    const int nBase    = mTile * 32;
    const int colBase0 = cGroup * 64;
    const int lmod = lane & 15;
    const int ldiv = lane >> 4;

    const unsigned short* aRow0 = adjbf + (size_t)(nBase + lmod) * N_;
    const unsigned short* aRow1 = adjbf + (size_t)(nBase + 16 + lmod) * N_;
    const int aOff0 = 16 * ldiv;
    const int aOff1 = 16 * ldiv + 8;

    const unsigned short* bRow0 = Ft + (size_t)(colBase0 +  0 + lmod) * N_ + 16 * ldiv;
    const unsigned short* bRow1 = Ft + (size_t)(colBase0 + 16 + lmod) * N_ + 16 * ldiv;
    const unsigned short* bRow2 = Ft + (size_t)(colBase0 + 32 + lmod) * N_ + 16 * ldiv;
    const unsigned short* bRow3 = Ft + (size_t)(colBase0 + 48 + lmod) * N_ + 16 * ldiv;

    v8f c00 = {}, c01 = {}, c02 = {}, c03 = {};
    v8f c10 = {}, c11 = {}, c12 = {}, c13 = {};

    for (int k = 0; k < N_; k += 32) {
        PackA a0, a1;
        a0.q[0] = *(const uint4*)(aRow0 + k + aOff0);
        a0.q[1] = *(const uint4*)(aRow0 + k + aOff1);
        a1.q[0] = *(const uint4*)(aRow1 + k + aOff0);
        a1.q[1] = *(const uint4*)(aRow1 + k + aOff1);

        PackA b0, b1, b2, b3;
        b0.q[0] = *(const uint4*)(bRow0 + k);     b0.q[1] = *(const uint4*)(bRow0 + k + 8);
        b1.q[0] = *(const uint4*)(bRow1 + k);     b1.q[1] = *(const uint4*)(bRow1 + k + 8);
        b2.q[0] = *(const uint4*)(bRow2 + k);     b2.q[1] = *(const uint4*)(bRow2 + k + 8);
        b3.q[0] = *(const uint4*)(bRow3 + k);     b3.q[1] = *(const uint4*)(bRow3 + k + 8);

        c00 = __builtin_amdgcn_wmma_f32_16x16x32_bf16(false, a0.v, false, b0.v, (short)0, c00, false, false);
        c01 = __builtin_amdgcn_wmma_f32_16x16x32_bf16(false, a0.v, false, b1.v, (short)0, c01, false, false);
        c02 = __builtin_amdgcn_wmma_f32_16x16x32_bf16(false, a0.v, false, b2.v, (short)0, c02, false, false);
        c03 = __builtin_amdgcn_wmma_f32_16x16x32_bf16(false, a0.v, false, b3.v, (short)0, c03, false, false);
        c10 = __builtin_amdgcn_wmma_f32_16x16x32_bf16(false, a1.v, false, b0.v, (short)0, c10, false, false);
        c11 = __builtin_amdgcn_wmma_f32_16x16x32_bf16(false, a1.v, false, b1.v, (short)0, c11, false, false);
        c12 = __builtin_amdgcn_wmma_f32_16x16x32_bf16(false, a1.v, false, b2.v, (short)0, c12, false, false);
        c13 = __builtin_amdgcn_wmma_f32_16x16x32_bf16(false, a1.v, false, b3.v, (short)0, c13, false, false);
    }

#pragma unroll
    for (int mt = 0; mt < 2; ++mt) {
        const int nOut = nBase + mt * 16 + 8 * ldiv;
#pragma unroll
        for (int t = 0; t < 4; ++t) {
            v8f c;
            if (mt == 0) c = (t == 0) ? c00 : (t == 1) ? c01 : (t == 2) ? c02 : c03;
            else         c = (t == 0) ? c10 : (t == 1) ? c11 : (t == 2) ? c12 : c13;
            float* out = Yt + (size_t)(colBase0 + t * 16 + lmod) * N_ + nOut;
            float4 lo = make_float4(c[0], c[1], c[2], c[3]);
            float4 hi = make_float4(c[4], c[5], c[6], c[7]);
            *(float4*)(out)     = lo;
            *(float4*)(out + 4) = hi;
        }
    }
}

// ---------------------------------------------------------------------------
// Pointwise PDE + GRU step: one thread per (b, n).
// ---------------------------------------------------------------------------
__global__ __launch_bounds__(256) void pde_step_kernel(
    const float* __restrict__ p1W, const float* __restrict__ p1B,
    const float* __restrict__ p2W, const float* __restrict__ p2B,
    const float* __restrict__ zW,  const float* __restrict__ zB,
    const float* __restrict__ hW,  const float* __restrict__ hB,
    const float* __restrict__ oW,  const float* __restrict__ oB,
    const float* __restrict__ pmix,
    float* __restrict__ field, float* __restrict__ state,
    const float* __restrict__ Yt, unsigned short* __restrict__ Ft) {
    const int idx = blockIdx.x * blockDim.x + threadIdx.x;
    const int b = idx >> 12;
    const int n = idx & (N_ - 1);
    const float alpha = sigmf(pmix[0]);
    const float dt = 1.0f / (float)STEPS_;

    float f[FD_], s[FD_], y[FD_];
    const size_t base = (size_t)idx * FD_;
#pragma unroll
    for (int k = 0; k < FD_; ++k) {
        f[k] = field[base + k];
        s[k] = state[base + k];
        y[k] = Yt[(size_t)(b * FD_ + k) * N_ + n];
    }

    // NF = pde2(tanh(pde1(f))), streamed over hidden dim
    float nf[FD_];
#pragma unroll
    for (int k = 0; k < FD_; ++k) nf[k] = p2B[k];
    for (int j = 0; j < HD_; ++j) {
        float a = p1B[j];
#pragma unroll
        for (int i = 0; i < FD_; ++i) a = fmaf(f[i], p1W[i * HD_ + j], a);
        a = tanhf(a);
#pragma unroll
        for (int k = 0; k < FD_; ++k) nf[k] = fmaf(a, p2W[j * FD_ + k], nf[k]);
    }

    float fe[FD_];
#pragma unroll
    for (int k = 0; k < FD_; ++k) {
        const float lf = y[k] - f[k];
        const float dF = alpha * lf + (1.0f - alpha) * nf[k];
        fe[k] = f[k] + dF * dt;
    }

    // GRU gates on fs = [fe, s]
    float zz[FD_], hh[FD_];
#pragma unroll
    for (int k = 0; k < FD_; ++k) { zz[k] = zB[k]; hh[k] = hB[k]; }
#pragma unroll
    for (int i = 0; i < 2 * FD_; ++i) {
        const float v = (i < FD_) ? fe[i] : s[i - FD_];
#pragma unroll
        for (int k = 0; k < FD_; ++k) {
            zz[k] = fmaf(v, zW[i * FD_ + k], zz[k]);
            hh[k] = fmaf(v, hW[i * FD_ + k], hh[k]);
        }
    }

    float ns[FD_];
#pragma unroll
    for (int k = 0; k < FD_; ++k) {
        const float z  = sigmf(zz[k]);
        const float hv = tanhf(hh[k]);
        ns[k] = (1.0f - z) * s[k] + z * hv;
    }

    float fr[FD_];
#pragma unroll
    for (int k = 0; k < FD_; ++k) fr[k] = fe[k] + oB[k];
#pragma unroll
    for (int i = 0; i < FD_; ++i) {
        const float v = ns[i];
#pragma unroll
        for (int k = 0; k < FD_; ++k) fr[k] = fmaf(v, oW[i * FD_ + k], fr[k]);
    }

#pragma unroll
    for (int k = 0; k < FD_; ++k) {
        field[base + k] = fr[k];
        state[base + k] = ns[k];
        Ft[(size_t)(b * FD_ + k) * N_ + n] = f2bf(fr[k]);
    }
}

// ---------------------------------------------------------------------------
// Decoder: one thread per (b, n); writes (B, OUT, N) layout.
// ---------------------------------------------------------------------------
__global__ __launch_bounds__(256) void decoder_kernel(
    const float* __restrict__ diW, const float* __restrict__ diB,
    const float* __restrict__ m1aW, const float* __restrict__ m1aB,
    const float* __restrict__ m1bW, const float* __restrict__ m1bB,
    const float* __restrict__ m2aW, const float* __restrict__ m2aB,
    const float* __restrict__ m2bW, const float* __restrict__ m2bB,
    const float* __restrict__ doW, const float* __restrict__ doB,
    const float* __restrict__ field, float* __restrict__ out) {
    const int idx = blockIdx.x * blockDim.x + threadIdx.x;
    const int b = idx >> 12;
    const int n = idx & (N_ - 1);

    float f[FD_];
    const size_t base = (size_t)idx * FD_;
#pragma unroll
    for (int k = 0; k < FD_; ++k) f[k] = field[base + k];

    float h[HD_];
#pragma unroll
    for (int j = 0; j < HD_; ++j) {
        float a = diB[j];
#pragma unroll
        for (int i = 0; i < FD_; ++i) a = fmaf(f[i], diW[i * HD_ + j], a);
        h[j] = a;
    }

    mlp64(h, m1aW, m1aB, m1bW, m1bB);
    mlp64(h, m2aW, m2aB, m2bW, m2bB);

    for (int o = 0; o < OUT_; ++o) {
        float a = doB[o];
#pragma unroll
        for (int i = 0; i < HD_; ++i) a = fmaf(h[i], doW[i * OUT_ + o], a);
        out[(size_t)(b * OUT_ + o) * N_ + n] = a;
    }
}

// ---------------------------------------------------------------------------
// Host launcher.  Input order assumes recursive insertion-order flattening of
// setup_inputs(): history, future, batch_seen, epoch, train, then params:
// enc_in.{W,b}, enc_m1.fc1.{W,b}, enc_m1.fc2.{W,b}, enc_m2.fc1.{W,b},
// enc_m2.fc2.{W,b}, enc_out.{W,b}, tid_emb, diw_emb, t2f.{W,b}, E1, E2,
// pde1.{W,b}, pde2.{W,b}, Wz.{W,b}, Wh.{W,b}, Wo.{W,b}, dec_in.{W,b},
// dec_m1.fc1.{W,b}, dec_m1.fc2.{W,b}, dec_m2.fc1.{W,b}, dec_m2.fc2.{W,b},
// dec_out.{W,b}, pde_mix.
// ---------------------------------------------------------------------------
extern "C" void kernel_launch(void* const* d_in, const int* in_sizes, int n_in,
                              void* d_out, int out_size, void* d_ws, size_t ws_size,
                              hipStream_t stream) {
    (void)in_sizes; (void)n_in; (void)out_size; (void)ws_size;
    #define IN(i) ((const float*)d_in[(i)])

    const float* hist  = IN(0);
    const float* eiW   = IN(5),  *eiB  = IN(6);
    const float* e1aW  = IN(7),  *e1aB = IN(8),  *e1bW = IN(9),  *e1bB = IN(10);
    const float* e2aW  = IN(11), *e2aB = IN(12), *e2bW = IN(13), *e2bB = IN(14);
    const float* eoW   = IN(15), *eoB  = IN(16);
    const float* tidE  = IN(17), *diwE = IN(18);
    const float* tfW   = IN(19), *tfB  = IN(20);
    const float* E1    = IN(21), *E2   = IN(22);
    const float* p1W   = IN(23), *p1B  = IN(24), *p2W = IN(25), *p2B = IN(26);
    const float* zW    = IN(27), *zB   = IN(28);
    const float* hW    = IN(29), *hB   = IN(30);
    const float* oW    = IN(31), *oB   = IN(32);
    const float* diW   = IN(33), *diB  = IN(34);
    const float* d1aW  = IN(35), *d1aB = IN(36), *d1bW = IN(37), *d1bB = IN(38);
    const float* d2aW  = IN(39), *d2aB = IN(40), *d2bW = IN(41), *d2bB = IN(42);
    const float* doW   = IN(43), *doB  = IN(44);
    const float* pmix  = IN(45);
    #undef IN

    // workspace layout
    char* ws = (char*)d_ws;
    unsigned short* adjbf = (unsigned short*)(ws);                          // 32 MB
    unsigned short* Ft    = (unsigned short*)(ws + (size_t)33554432);       //  8 MB
    float*          Yt    = (float*)(ws + (size_t)41943040);                // 16 MB
    float*          field = (float*)(ws + (size_t)58720256);                // 16 MB
    float*          state = (float*)(ws + (size_t)75497472);                // 16 MB

    const int PTS = B_ * N_;            // 131072 per-node threads
    const dim3 ptBlk(256), ptGrd(PTS / 256);

    // 1) adjacency softmax -> bf16
    adj_kernel<<<N_, 256, 0, stream>>>(E1, E2, adjbf);

    // 2) encoder
    encoder_kernel<<<ptGrd, ptBlk, 0, stream>>>(hist, eiW, eiB,
                                                e1aW, e1aB, e1bW, e1bB,
                                                e2aW, e2aB, e2bW, e2bB,
                                                eoW, eoB, tidE, diwE, tfW, tfB,
                                                field, state, Ft);

    // 3) PDE scan: 4x (WMMA GEMM + pointwise update)
    const dim3 gmBlk(128), gmGrd(512);   // 2048 waves, 32x64 output strip each
    for (int s = 0; s < STEPS_; ++s) {
        gemm_adj<<<gmGrd, gmBlk, 0, stream>>>(adjbf, Ft, Yt);
        pde_step_kernel<<<ptGrd, ptBlk, 0, stream>>>(p1W, p1B, p2W, p2B,
                                                     zW, zB, hW, hB, oW, oB,
                                                     pmix, field, state, Yt, Ft);
    }

    // 4) decoder -> d_out (B, OUT, N, 1) f32
    decoder_kernel<<<ptGrd, ptBlk, 0, stream>>>(diW, diB,
                                                d1aW, d1aB, d1bW, d1bB,
                                                d2aW, d2aB, d2bW, d2bB,
                                                doW, doB, field, (float*)d_out);
}
